// NODE_48507360641641
// MI455X (gfx1250) — matrix operand order repaired and checked by reading, not compiled, not run on hardware
//
#include <hip/hip_runtime.h>

// ---------------------------------------------------------------------------
// Neural ODE (10 blocks x 16 DOPRI5 steps x 6 drift evals of a
// 30->1024->1024->30 LeakyReLU MLP) on gfx1250 via bf16 WMMA.
// ---------------------------------------------------------------------------

typedef __bf16 v16bf __attribute__((ext_vector_type(16)));
typedef float  v8f   __attribute__((ext_vector_type(8)));
typedef unsigned int u32x4 __attribute__((ext_vector_type(4)));

union BFrag { v16bf v; u32x4 q[2]; };

#define HID   1024
#define LOOK  30
#define DEPTH 10
#define STEPS 16
#define HSTR  1032   // bf16 LDS activation row stride (padded: bank-conflict free b128)
#define ASTR  40     // bf16 LDS ytmp row stride (padded)

__device__ __forceinline__ unsigned short f2bf(float f) {
    unsigned int u = __float_as_uint(f);
    u += 0x7fffu + ((u >> 16) & 1u);      // round-to-nearest-even
    return (unsigned short)(u >> 16);
}

__device__ __forceinline__ v8f wmma_bf16(v16bf a, v16bf b, v8f c) {
    return __builtin_amdgcn_wmma_f32_16x16x32_bf16(false, a, false, b,
                                                   (short)0, c, false, false);
}

// --------------------------- weight repack ---------------------------------
// W1 [10][30][1024] f32 -> W1t [10][1024][32]  bf16 (N-major, K padded 30->32)
// W2 [10][1024][1024]   -> W2t [10][1024][1024] bf16 (N-major)
// W3 [10][1024][30]     -> W3t [10][32][1024]  bf16 (N-major, N padded 30->32)
__global__ void convert_weights(const float* __restrict__ W1,
                                const float* __restrict__ W2,
                                const float* __restrict__ W3,
                                unsigned short* __restrict__ W1t,
                                unsigned short* __restrict__ W2t,
                                unsigned short* __restrict__ W3t) {
    const long long N1 = (long long)DEPTH * HID * 32;
    const long long N2 = (long long)DEPTH * HID * HID;
    const long long N3 = (long long)DEPTH * 32 * HID;
    long long i = (long long)blockIdx.x * blockDim.x + threadIdx.x;
    const long long stride = (long long)gridDim.x * blockDim.x;
    for (; i < N1 + N2 + N3; i += stride) {
        if (i < N1) {
            long long t = i;
            int k = (int)(t & 31);  t >>= 5;
            int n = (int)(t & 1023);
            int d = (int)(t >> 10);
            float v = (k < LOOK) ? W1[((long long)d * LOOK + k) * HID + n] : 0.f;
            W1t[i] = f2bf(v);
        } else if (i < N1 + N2) {
            long long j = i - N1;
            int k = (int)(j & 1023);
            int n = (int)((j >> 10) & 1023);
            int d = (int)(j >> 20);
            W2t[j] = f2bf(W2[((long long)d * HID + k) * HID + n]);
        } else {
            long long j = i - N1 - N2;
            int k = (int)(j & 1023);
            int n = (int)((j >> 10) & 31);
            int d = (int)(j >> 15);
            float v = (n < LOOK) ? W3[((long long)d * HID + k) * LOOK + n] : 0.f;
            W3t[j] = f2bf(v);
        }
    }
}

// ----------------------------- ODE solver ----------------------------------
// 512 blocks x 256 threads (8 wave32). Block owns 32 batch rows.
// (256, 4): cap VGPRs at 256/wave -> 4 waves/SIMD, 2 blocks/WGP (LDS allows it).
__launch_bounds__(256, 4)
__global__ void node_solve(const float* __restrict__ y0,
                           const float* __restrict__ b1g,
                           const float* __restrict__ b2g,
                           const float* __restrict__ b3g,
                           const float* __restrict__ Wf,
                           const float* __restrict__ bfp,
                           const unsigned short* __restrict__ W1t,
                           const unsigned short* __restrict__ W2t,
                           const unsigned short* __restrict__ W3t,
                           float* __restrict__ out) {
    extern __shared__ char smem[];
    float*          yL  = (float*)smem;                          //  32*32 f32 (4 KB)
    float*          kL  = yL + 32 * 32;                          // 6*32*32 f32 (24 KB)
    float*          b1L = kL + 6 * 32 * 32;                      // 1024 f32 (4 KB)
    float*          b2L = b1L + HID;                             // 1024 f32 (4 KB)
    float*          b3L = b2L + HID;                             //   32 f32 (128 B)
    unsigned short* ast = (unsigned short*)(b3L + 32);           //  32*ASTR bf16 (2.5 KB)
    unsigned short* hb  = ast + 32 * ASTR;                       //  32*HSTR bf16 (64.5 KB)

    const int tid  = threadIdx.x;
    const int lane = tid & 31;
    const int wave = tid >> 5;
    const int half = lane >> 4;   // 0: lanes 0-15, 1: lanes 16-31
    const int l16  = lane & 15;
    const int rowbase = blockIdx.x * 32;

    // DOPRI5 tableau
    static const float AC[6][5] = {
        {0.f, 0.f, 0.f, 0.f, 0.f},
        {1.f / 5.f, 0.f, 0.f, 0.f, 0.f},
        {3.f / 40.f, 9.f / 40.f, 0.f, 0.f, 0.f},
        {44.f / 45.f, -56.f / 15.f, 32.f / 9.f, 0.f, 0.f},
        {19372.f / 6561.f, -25360.f / 2187.f, 64448.f / 6561.f, -212.f / 729.f, 0.f},
        {9017.f / 3168.f, -355.f / 33.f, 46732.f / 5247.f, 49.f / 176.f, -5103.f / 18656.f}};
    static const float BC[6] = {35.f / 384.f, 0.f, 500.f / 1113.f,
                                125.f / 192.f, -2187.f / 6784.f, 11.f / 84.f};
    const float hstep = 1.0f / (float)STEPS;

    // load y0 tile
    for (int i = tid; i < 32 * LOOK; i += 256) {
        int m = i / LOOK, j = i % LOOK;
        yL[m * 32 + j] = y0[(long long)(rowbase + m) * LOOK + j];
    }
    __syncthreads();

    for (int d = 0; d < DEPTH; ++d) {
        const unsigned short* W1d = W1t + (long long)d * HID * 32;
        const unsigned short* W2d = W2t + (long long)d * HID * HID;
        const unsigned short* W3d = W3t + (long long)d * 32 * HID;

        // per-depth biases -> LDS (keeps them out of long-lived VGPRs;
        // barriers prevent LICM from re-hoisting the ds loads)
        for (int i = tid; i < HID; i += 256) {
            b1L[i] = b1g[d * HID + i];
            b2L[i] = b2g[d * HID + i];
        }
        if (tid < 32) b3L[tid] = (tid < LOOK) ? b3g[d * LOOK + tid] : 0.f;
        __syncthreads();

        for (int st = 0; st < STEPS; ++st) {
            for (int s = 0; s < 6; ++s) {
                // ---- stage input: ytmp = y + h*sum a[s][p]*k_p  -> bf16 LDS
                for (int i = tid; i < 32 * 32; i += 256) {
                    int m = i >> 5, j = i & 31;
                    float v = 0.f;
                    if (j < LOOK) {
                        v = yL[m * 32 + j];
                        for (int p = 0; p < s; ++p)
                            v += hstep * AC[s][p] * kL[(p * 32 + m) * 32 + j];
                    }
                    ast[m * ASTR + j] = f2bf(v);
                }
                __syncthreads();

                // ---- layer 1: h1 = leaky(ytmp @ W1 + b1); bias folded into C
                {
                    BFrag a[2];
#pragma unroll
                    for (int mt = 0; mt < 2; ++mt) {
                        const unsigned short* ap =
                            ast + (l16 + 16 * mt) * ASTR + half * 8;
                        a[mt].q[0] = *(const u32x4*)ap;
                        a[mt].q[1] = *(const u32x4*)(ap + 16);
                    }
#pragma unroll
                    for (int nt = 0; nt < 8; ++nt) {
                        const int col = wave * 128 + nt * 16 + l16;
                        BFrag b;
                        const unsigned short* bp = W1d + col * 32 + half * 8;
                        b.q[0] = *(const u32x4*)bp;
                        b.q[1] = *(const u32x4*)(bp + 16);
                        const float bv = b1L[col];
#pragma unroll
                        for (int mt = 0; mt < 2; ++mt) {
                            v8f acc;
#pragma unroll
                            for (int r = 0; r < 8; ++r) acc[r] = bv;
                            acc = wmma_bf16(a[mt].v, b.v, acc);
#pragma unroll
                            for (int r = 0; r < 8; ++r) {
                                float v = fmaxf(acc[r], 0.3f * acc[r]); // LeakyReLU
                                hb[(r + 8 * half + 16 * mt) * HSTR + col] = f2bf(v);
                            }
                        }
                    }
                }
                __syncthreads();

                // ---- layer 2: h2 = leaky(h1 @ W2 + b2), K = 1024
                {
                    v8f acc[2][8];
#pragma unroll
                    for (int nt = 0; nt < 8; ++nt) {
                        const float bv = b2L[wave * 128 + nt * 16 + l16];
#pragma unroll
                        for (int mt = 0; mt < 2; ++mt)
#pragma unroll
                            for (int r = 0; r < 8; ++r) acc[mt][nt][r] = bv;
                    }

#pragma unroll 1
                    for (int ko = 0; ko < HID; ko += 32) {
                        BFrag a[2];
#pragma unroll
                        for (int mt = 0; mt < 2; ++mt) {
                            const unsigned short* ap =
                                hb + (l16 + 16 * mt) * HSTR + ko + half * 8;
                            a[mt].q[0] = *(const u32x4*)ap;
                            a[mt].q[1] = *(const u32x4*)(ap + 16);
                        }
                        // two N-groups of 4: halves peak live B-fragment regs
#pragma unroll
                        for (int ng = 0; ng < 2; ++ng) {
#pragma unroll
                            for (int n4 = 0; n4 < 4; ++n4) {
                                const int nt = ng * 4 + n4;
                                const int col = wave * 128 + nt * 16 + l16;
                                const unsigned short* bp =
                                    W2d + (long long)col * HID + ko + half * 8;
                                BFrag b;
                                b.q[0] = *(const u32x4*)bp;
                                b.q[1] = *(const u32x4*)(bp + 16);
#pragma unroll
                                for (int mt = 0; mt < 2; ++mt)
                                    acc[mt][nt] = wmma_bf16(a[mt].v, b.v, acc[mt][nt]);
                            }
                            __builtin_amdgcn_sched_barrier(0);
                        }
                    }
                    __syncthreads();   // all waves done reading h1
#pragma unroll
                    for (int nt = 0; nt < 8; ++nt) {
                        const int col = wave * 128 + nt * 16 + l16;
#pragma unroll
                        for (int mt = 0; mt < 2; ++mt)
#pragma unroll
                            for (int r = 0; r < 8; ++r) {
                                float v = acc[mt][nt][r];
                                v = fmaxf(v, 0.3f * v);
                                hb[(r + 8 * half + 16 * mt) * HSTR + col] = f2bf(v);
                            }
                    }
                }
                __syncthreads();

                // ---- layer 3: k_s = h2 @ W3 + b3 (waves 0..3; wave-uniform)
                if (wave < 4) {
                    const int mt = wave & 1, nt = wave >> 1;
                    const int col = nt * 16 + l16;
                    const float bv3 = b3L[col];   // 0 in padded cols
                    v8f acc;
#pragma unroll
                    for (int r = 0; r < 8; ++r) acc[r] = bv3;
#pragma unroll 1
                    for (int ko = 0; ko < HID; ko += 32) {
                        BFrag a, b;
                        const unsigned short* ap =
                            hb + (l16 + 16 * mt) * HSTR + ko + half * 8;
                        a.q[0] = *(const u32x4*)ap;
                        a.q[1] = *(const u32x4*)(ap + 16);
                        const unsigned short* bp =
                            W3d + (long long)col * HID + ko + half * 8;
                        b.q[0] = *(const u32x4*)bp;
                        b.q[1] = *(const u32x4*)(bp + 16);
                        acc = wmma_bf16(a.v, b.v, acc);
                        __builtin_amdgcn_sched_barrier(0);
                    }
                    if (col < LOOK) {
#pragma unroll
                        for (int r = 0; r < 8; ++r)
                            kL[(s * 32 + (r + 8 * half + 16 * mt)) * 32 + col] =
                                acc[r];
                    }
                }
                __syncthreads();
            } // stage s

            // ---- 5th-order update: y += h * sum b_p * k_p
            for (int i = tid; i < 32 * LOOK; i += 256) {
                int m = i / LOOK, j = i % LOOK;
                float v = yL[m * 32 + j];
#pragma unroll
                for (int p = 0; p < 6; ++p)
                    v += hstep * BC[p] * kL[(p * 32 + m) * 32 + j];
                yL[m * 32 + j] = v;
            }
            __syncthreads();
        } // step
    } // depth

    // ---- readout: out = y @ Wf + bf
    if (tid < 32) {
        float sacc = bfp[0];
        for (int j = 0; j < LOOK; ++j) sacc += yL[tid * 32 + j] * Wf[j];
        out[rowbase + tid] = sacc;
    }
}

// ---------------------------------------------------------------------------
extern "C" void kernel_launch(void* const* d_in, const int* in_sizes, int n_in,
                              void* d_out, int out_size, void* d_ws, size_t ws_size,
                              hipStream_t stream) {
    (void)in_sizes; (void)n_in; (void)out_size; (void)ws_size;
    const float* y0 = (const float*)d_in[0];
    const float* W1 = (const float*)d_in[1];
    const float* b1 = (const float*)d_in[2];
    const float* W2 = (const float*)d_in[3];
    const float* b2 = (const float*)d_in[4];
    const float* W3 = (const float*)d_in[5];
    const float* b3 = (const float*)d_in[6];
    const float* Wf = (const float*)d_in[7];
    const float* bf = (const float*)d_in[8];
    float* out = (float*)d_out;

    unsigned short* W1t = (unsigned short*)d_ws;                 // 10*1024*32
    unsigned short* W2t = W1t + (long long)DEPTH * HID * 32;     // 10*1024*1024
    unsigned short* W3t = W2t + (long long)DEPTH * HID * HID;    // 10*32*1024

    const long long total = (long long)DEPTH * HID * 32 +
                            (long long)DEPTH * HID * HID +
                            (long long)DEPTH * 32 * HID;
    const int cgrid = (int)((total + 255) / 256);
    convert_weights<<<cgrid, 256, 0, stream>>>(W1, W2, W3, W1t, W2t, W3t);

    const size_t shmem = (size_t)(32 * 32 + 6 * 32 * 32 + HID + HID + 32) * sizeof(float) +
                         (size_t)32 * ASTR * sizeof(unsigned short) +
                         (size_t)32 * HSTR * sizeof(unsigned short); // ~105.7 KB
    node_solve<<<dim3(16384 / 32), dim3(256), shmem, stream>>>(
        y0, b1, b2, b3, Wf, bf, W1t, W2t, W3t, out);
}
